// BackwardProjectionLite_16097537425486
// MI455X (gfx1250) — matrix operand order repaired and checked by reading, not compiled, not run on hardware
//
#include <hip/hip_runtime.h>

// ---------------------------------------------------------------------------
// BackwardProjectionLite for MI455X (gfx1250, wave32, WMMA)
//   P0: tiled transposes (bev -> pixel-major, context -> channel-last)
//   W : conv weight reorder OIHW -> [o][tap*256+ci] bf16
//   P2: projection + bilinear sampling + depth gating + gated residual fuse
//       -> fusedT [10000][256] bf16 (channel-last, WMMA B-operand layout ready)
//   D : implicit-GEMM 3x3 conv via v_wmma_f32_16x16x32_bf16 + BN + ReLU
//       (2 N-tiles per wave: one A fragment feeds two accumulators ->
//        halves weight L2 traffic, 144 wmma per wave)
// ---------------------------------------------------------------------------

typedef __attribute__((ext_vector_type(8)))  float  v8f;
typedef __attribute__((ext_vector_type(16))) __bf16 v16bf;
typedef __attribute__((ext_vector_type(8)))  __bf16 v8bf;

#define EMBED 256
#define BEVH  100
#define BEVW  100
#define NPIX  (BEVH * BEVW)      // 10000
#define ZA    4
#define NCAM  6
#define NREC  (NCAM * ZA)        // 24
#define DBINS 64
#define FH    32
#define FW    88
#define FSPAT (FH * FW)          // 2816
#define KTOT  (9 * EMBED)        // 2304
#define NTILES (NPIX / 16)       // 625

// ---------------- P0: generic 32x32 tiled transpose ----------------
// src [batch][rows][cols] -> dst [batch][cols][rows]
__global__ void transpose_f32(const float* __restrict__ src, float* __restrict__ dst,
                              int rows, int cols) {
  __shared__ float tile[32][33];
  const size_t bo = (size_t)blockIdx.z * rows * cols;
  src += bo; dst += bo;
  const int c0 = blockIdx.x * 32, r0 = blockIdx.y * 32;
  for (int i = threadIdx.y; i < 32; i += 8) {
    int r = r0 + i, c = c0 + threadIdx.x;
    tile[i][threadIdx.x] = (r < rows && c < cols) ? src[(size_t)r * cols + c] : 0.0f;
  }
  __syncthreads();
  for (int i = threadIdx.y; i < 32; i += 8) {
    int c = c0 + i, r = r0 + threadIdx.x;
    if (c < cols && r < rows) dst[(size_t)c * rows + r] = tile[threadIdx.x][i];
  }
}

// ---------------- W: weight reorder to WMMA A-operand friendly layout -------
// wT[o][t*256 + i] = conv_w[o][i][t/3][t%3]  (bf16)
__global__ void weight_reorder(const float* __restrict__ w, __bf16* __restrict__ wT) {
  int tid = blockIdx.x * 256 + threadIdx.x;
  if (tid >= EMBED * KTOT) return;
  int o = tid / KTOT;
  int k = tid - o * KTOT;
  int t = k >> 8;            // tap 0..8
  int i = k & 255;           // in channel
  int ky = t / 3, kx = t - ky * 3;
  wT[tid] = (__bf16)w[(((size_t)o * EMBED + i) * 3 + ky) * 3 + kx];
}

// ---------------- P2: projection + sampling + gated residual fusion ---------
// one block per BEV pixel; threads = channels
__global__ __launch_bounds__(256) void sample_fuse(
    const float* __restrict__ bevT,     // [10000][256]
    const float* __restrict__ ctxT,     // [6][2816][256]
    const float* __restrict__ dpr,      // [6][64][32][88]
    const float* __restrict__ l2i,      // [6][4][4]
    const float* __restrict__ imhw,     // [6][2]
    __bf16* __restrict__ fusedT) {      // [10000][256]
  __shared__ float s_w[NREC][4];
  __shared__ int   s_off[NREC][4];
  __shared__ float s_prob[NREC];

  const int p  = blockIdx.x;
  const int hy = p / BEVW;
  const int wx = p - hy * BEVW;

  if (threadIdx.x < NREC) {
    const int r   = threadIdx.x;
    const int cam = r >> 2;
    const int z   = r & 3;
    // reference anchors: x along W, y along H, z anchors
    const float X = -51.2f + 1.024f * ((float)wx + 0.5f);
    const float Y = -51.2f + 1.024f * ((float)hy + 0.5f);
    const float Z = -5.0f + 2.0f * ((float)z + 0.5f);
    const float* M = l2i + cam * 16;
    const float px = M[0] * X + M[1] * Y + M[2]  * Z + M[3];
    const float py = M[4] * X + M[5] * Y + M[6]  * Z + M[7];
    const float pd = M[8] * X + M[9] * Y + M[10] * Z + M[11];
    const float dcl = fmaxf(pd, 1e-5f);
    const float imh = fmaxf(imhw[cam * 2 + 0], 1.0f);
    const float imw = fmaxf(imhw[cam * 2 + 1], 1.0f);
    const float xn = (px / dcl) / imw;
    const float yn = (py / dcl) / imh;
    const bool mask = (pd > 1e-5f) && (xn > 1e-5f) && (xn < 1.0f - 1e-5f) &&
                      (yn > 1e-5f) && (yn < 1.0f - 1e-5f);
    // grid_sample coords (align_corners=False) on the 32x88 feature map
    const float xp = xn * (float)FW - 0.5f;
    const float yp = yn * (float)FH - 0.5f;
    const float x0f = floorf(xp), y0f = floorf(yp);
    const float wx1 = xp - x0f, wx0 = 1.0f - wx1;
    const float wy1 = yp - y0f, wy0 = 1.0f - wy1;
    const int x0 = (int)x0f, y0 = (int)y0f;
    int bin = (int)rintf((pd - 1.0f) * (float)(DBINS - 1) / 59.0f);
    bin = min(max(bin, 0), DBINS - 1);

    float tw[4] = {wx0 * wy0, wx1 * wy0, wx0 * wy1, wx1 * wy1};
    const int tx[4] = {x0, x0 + 1, x0, x0 + 1};
    const int ty[4] = {y0, y0, y0 + 1, y0 + 1};
    float prob = 0.0f;
#pragma unroll
    for (int t = 0; t < 4; ++t) {
      const bool v = (tx[t] >= 0) && (tx[t] <= FW - 1) && (ty[t] >= 0) && (ty[t] <= FH - 1);
      const int cx = min(max(tx[t], 0), FW - 1);
      const int cy = min(max(ty[t], 0), FH - 1);
      if (mask && v)
        prob += tw[t] * dpr[(((size_t)cam * DBINS + bin) * FH + cy) * FW + cx];
      tw[t] = v ? tw[t] : 0.0f;
      s_off[r][t] = (cam * FSPAT + cy * FW + cx) * EMBED;
    }
    if (!mask) prob = 0.0f;
    s_prob[r] = prob;
#pragma unroll
    for (int t = 0; t < 4; ++t) s_w[r][t] = prob * tw[t];
  }
  __syncthreads();

  const int c = threadIdx.x;
  float wsum = 0.0f;
#pragma unroll
  for (int r = 0; r < NREC; ++r) wsum += s_prob[r];
  float acc = 0.0f;
  for (int r = 0; r < NREC; ++r) {
    if (s_prob[r] != 0.0f) {          // uniform across the block (LDS broadcast)
#pragma unroll
      for (int t = 0; t < 4; ++t)
        acc += s_w[r][t] * ctxT[s_off[r][t] + c];   // coalesced over channels
    }
  }
  const float ctx_bev = acc / fmaxf(wsum, 1e-6f);
  const float dw = fminf(fmaxf(wsum * (1.0f / (float)NREC), 0.0f), 1.0f);
  const float fused = bevT[(size_t)p * EMBED + c] + ctx_bev * dw;
  fusedT[(size_t)p * EMBED + c] = (__bf16)fused;
}

// ---------------- D: implicit-GEMM 3x3 conv via WMMA bf16 + BN + ReLU -------
// M = out channels (16 tiles), N = pixels (625 tiles), K = 2304 (72 x 32)
// grid = (313, 2), block = 256 (8 waves); wave w -> m_tile = blockIdx.y*8 + w
// Each wave processes TWO adjacent N-tiles with a single shared A fragment.
__global__ __launch_bounds__(256) void conv_wmma(
    const __bf16* __restrict__ fusedT,  // [10000][256]
    const __bf16* __restrict__ wT,      // [256][2304]
    const float* __restrict__ gmm,      // bn_gamma
    const float* __restrict__ bta,      // bn_beta
    const float* __restrict__ mea,      // bn_mean
    const float* __restrict__ var,      // bn_var
    float* __restrict__ out) {          // [256][10000]
  const int wave   = threadIdx.x >> 5;
  const int lane   = threadIdx.x & 31;
  const int half   = lane >> 4;
  const int ln     = lane & 15;
  const int m_tile = blockIdx.y * 8 + wave;

  const int n0 = blockIdx.x * 2;                      // first pixel tile
  const bool n1_valid = (n0 + 1) < NTILES;            // wave-uniform
  const int n1 = n1_valid ? (n0 + 1) : (NTILES - 1);  // clamp: duplicate compute

  const int o = m_tile * 16 + ln;                     // A-row: out channel
  const __bf16* __restrict__ wrow = wT + (size_t)o * KTOT;

  // per-tile pixel -> 9 shifted-neighbor offsets with border zero-select
  int  poff0[9], poff1[9];
  bool pval0[9], pval1[9];
  {
    const int p0 = n0 * 16 + ln;
    const int p1 = n1 * 16 + ln;
    const int y0 = p0 / BEVW, x0 = p0 - y0 * BEVW;
    const int y1 = p1 / BEVW, x1 = p1 - y1 * BEVW;
#pragma unroll
    for (int t = 0; t < 9; ++t) {
      const int dy = t / 3 - 1, dx = t % 3 - 1;
      int yy = y0 + dy, xx = x0 + dx;
      bool v = (yy >= 0) && (yy < BEVH) && (xx >= 0) && (xx < BEVW);
      pval0[t] = v;
      poff0[t] = v ? (yy * BEVW + xx) * EMBED : 0;
      yy = y1 + dy; xx = x1 + dx;
      v = (yy >= 0) && (yy < BEVH) && (xx >= 0) && (xx < BEVW);
      pval1[t] = v;
      poff1[t] = v ? (yy * BEVW + xx) * EMBED : 0;
    }
  }

  v16bf zerob;
#pragma unroll
  for (int i = 0; i < 16; ++i) zerob[i] = (__bf16)0.0f;
  v8f acc0 = {};
  v8f acc1 = {};

  for (int t = 0; t < 9; ++t) {
    const __bf16* __restrict__ abase  = wrow + t * EMBED;
    const __bf16* __restrict__ bbase0 = fusedT + poff0[t] + 16 * half;
    const __bf16* __restrict__ bbase1 = fusedT + poff1[t] + 16 * half;
    const bool bv0 = pval0[t];
    const bool bv1 = pval1[t];
    if (t < 8) __builtin_prefetch(wrow + (t + 1) * EMBED, 0, 0);
#pragma unroll
    for (int c8 = 0; c8 < 8; ++c8) {
      const int ci0 = c8 * 32;
      // A 16x32 bf16 frag: k(e) = e + 16*half -> two 16B loads
      union { v16bf v; v8bf h[2]; } af;
      af.h[0] = *(const v8bf*)(abase + ci0 + 16 * half);
      af.h[1] = *(const v8bf*)(abase + ci0 + 8 + 16 * half);
      // B 32x16 bf16 frags: k(e) = e + 16*half -> one contiguous 32B load each
      v16bf bf0 = *(const v16bf*)(bbase0 + ci0);
      v16bf bf1 = *(const v16bf*)(bbase1 + ci0);
      bf0 = bv0 ? bf0 : zerob;
      bf1 = bv1 ? bf1 : zerob;
      acc0 = __builtin_amdgcn_wmma_f32_16x16x32_bf16(
          false, af.v, false, bf0, (short)0, acc0, false, false);
      acc1 = __builtin_amdgcn_wmma_f32_16x16x32_bf16(
          false, af.v, false, bf1, (short)0, acc1, false, false);
    }
  }

  // fused BN + ReLU epilogue; D layout: VGPR r -> (ch = r + 8*half), col = ln
  const int p0 = n0 * 16 + ln;
  const int p1 = n1 * 16 + ln;
#pragma unroll
  for (int r = 0; r < 8; ++r) {
    const int ch = m_tile * 16 + r + 8 * half;
    const float inv  = gmm[ch] * rsqrtf(var[ch] + 1e-5f);
    const float bias = bta[ch] - mea[ch] * inv;
    out[(size_t)ch * NPIX + p0] = fmaxf(acc0[r] * inv + bias, 0.0f);
    if (n1_valid)
      out[(size_t)ch * NPIX + p1] = fmaxf(acc1[r] * inv + bias, 0.0f);
  }
}

// ---------------------------------------------------------------------------
extern "C" void kernel_launch(void* const* d_in, const int* in_sizes, int n_in,
                              void* d_out, int out_size, void* d_ws, size_t ws_size,
                              hipStream_t stream) {
  const float* bev     = (const float*)d_in[0];   // [1,256,100,100]
  const float* context = (const float*)d_in[1];   // [1,6,256,32,88]
  const float* dprob   = (const float*)d_in[2];   // [1,6,64,32,88]
  const float* l2i     = (const float*)d_in[3];   // [1,6,4,4]
  const float* imhw    = (const float*)d_in[4];   // [1,6,2]
  const float* conv_w  = (const float*)d_in[5];   // [256,256,3,3]
  const float* bn_g    = (const float*)d_in[6];
  const float* bn_b    = (const float*)d_in[7];
  const float* bn_m    = (const float*)d_in[8];
  const float* bn_v    = (const float*)d_in[9];
  float* out = (float*)d_out;

  // workspace layout (bytes)
  char* ws = (char*)d_ws;
  float*  ctxT   = (float*)(ws);                                     // 6*2816*256*4 = 17,301,504
  float*  bevT   = (float*)(ws + 17301504);                          // 10000*256*4 = 10,240,000
  __bf16* fusedT = (__bf16*)(ws + 17301504 + 10240000);              // 10000*256*2 =  5,120,000
  __bf16* wT     = (__bf16*)(ws + 17301504 + 10240000 + 5120000);    // 256*2304*2  =  1,179,648
  (void)in_sizes; (void)n_in; (void)out_size; (void)ws_size;

  // P0a: bev [256][10000] -> bevT [10000][256]
  {
    dim3 blk(32, 8), grd((NPIX + 31) / 32, (EMBED + 31) / 32, 1);
    transpose_f32<<<grd, blk, 0, stream>>>(bev, bevT, EMBED, NPIX);
  }
  // P0b: context per cam [256][2816] -> ctxT [cam][2816][256]
  {
    dim3 blk(32, 8), grd((FSPAT + 31) / 32, (EMBED + 31) / 32, NCAM);
    transpose_f32<<<grd, blk, 0, stream>>>(context, ctxT, EMBED, FSPAT);
  }
  // W: weight reorder -> bf16
  {
    int total = EMBED * KTOT;
    weight_reorder<<<(total + 255) / 256, 256, 0, stream>>>(conv_w, wT);
  }
  // P2: sampling + fusion
  sample_fuse<<<NPIX, 256, 0, stream>>>(bevT, ctxT, dprob, l2i, imhw, fusedT);
  // D: WMMA conv + BN + ReLU (two N-tiles per wave)
  {
    dim3 grd((NTILES + 1) / 2, 2);
    conv_wmma<<<grd, 256, 0, stream>>>(fusedT, wT, bn_g, bn_b, bn_m, bn_v, out);
  }
}
